// FeaturePreservationProjection_88304527606080
// MI455X (gfx1250) — matrix operand order repaired and checked by reading, not compiled, not run on hardware
//
#include <hip/hip_runtime.h>
#include <hip/hip_bf16.h>
#include <math.h>

// ---------------- problem constants (from reference) ----------------
#define BATCH      16384
#define DIM        4096
#define NUM_FACETS 64
#define TOPK       16          // NUM_FACETS / 4
#define MAX_ORDER  3
#define DELTA_MIN  0.01f
#define DELTA_MAX  0.5f
#define H_STEP     0.01f

typedef __attribute__((ext_vector_type(16))) _Float16 v16h;
typedef __attribute__((ext_vector_type(8)))  _Float16 v8h;
typedef __attribute__((ext_vector_type(8)))  float    v8f;

// ---------------- workspace layout (bytes) ----------------
// [0, 512K)        : normals in f16, row-major [64][4096]
// [512K, 512K+4M)  : proj f32, row-major [16384][64]
// then: abssum[64] f32, topidx[16] i32, deltas[16] f32, nn[16] f32
#define WS_NF16   0
#define WS_PROJ   (NUM_FACETS * DIM * 2)                      // 524288
#define WS_ABSSUM (WS_PROJ + (size_t)BATCH * NUM_FACETS * 4)  // +4MB
#define WS_TOPIDX (WS_ABSSUM + NUM_FACETS * 4)
#define WS_DELTAS (WS_TOPIDX + TOPK * 4)
#define WS_NN     (WS_DELTAS + TOPK * 4)

#define OUT_Q_ELEMS ((size_t)MAX_ORDER * BATCH * TOPK)        // 786432
#define OUT_E_OFF   OUT_Q_ELEMS                               // energy scalar
#define OUT_IDX_OFF (OUT_Q_ELEMS + 1)                         // 16 int32 slots

// =====================================================================
// K0: zero the accumulators we add into (ws/out are NOT re-poisoned
// between timed replays, so re-zero every launch).
// =====================================================================
__global__ void fpp_zero(float* abssum, float* energy_slot) {
    int t = threadIdx.x;
    if (t < NUM_FACETS) abssum[t] = 0.0f;
    if (t == NUM_FACETS) *energy_slot = 0.0f;
}

// =====================================================================
// K1: normalize each facet row (as the reference does) and emit an f16
// copy for the WMMA B operand. 64 blocks x 256 threads.
// =====================================================================
__global__ void fpp_normalize(const float* __restrict__ fn,
                              _Float16* __restrict__ nf16) {
    __shared__ float red[256];
    const int f = blockIdx.x;
    const int t = threadIdx.x;
    const float* row = fn + (size_t)f * DIM;

    float s = 0.0f;
    for (int i = t; i < DIM; i += 256) { float v = row[i]; s += v * v; }
    red[t] = s;
    __syncthreads();
    for (int o = 128; o > 0; o >>= 1) {
        if (t < o) red[t] += red[t + o];
        __syncthreads();
    }
    const float inv_norm = 1.0f / sqrtf(red[0]);

    _Float16* out = nf16 + (size_t)f * DIM;
    for (int i = t; i < DIM; i += 256)
        out[i] = (_Float16)(row[i] * inv_norm);
}

// =====================================================================
// WMMA fragment loaders.
// Operand layouts per CDNA5 ISA 7.12.2 (16-bit, wave32):
//   A (16x32): lane L -> m=L&15, half=L>>4
//     elems 0..7  = K(half*8 + e), elems 8..15 = K(16 + half*8 + e-8)
//   B (32x16): lane L -> n=L&15, half=L>>4; same K order as A
//   C/D (16x16 f32): elem r -> row (half*8 + r), col (L&15)
// `ap` must already include (+ half*8) and point at this K-step;
// `bp` must already include (+ half*8).
// =====================================================================
__device__ __forceinline__ v16h load_a_frag(const float* __restrict__ ap) {
    float4 a0 = *(const float4*)(ap + 0);
    float4 a1 = *(const float4*)(ap + 4);
    float4 a2 = *(const float4*)(ap + 16);
    float4 a3 = *(const float4*)(ap + 20);
    v16h A;
    A[0]  = (_Float16)a0.x; A[1]  = (_Float16)a0.y;
    A[2]  = (_Float16)a0.z; A[3]  = (_Float16)a0.w;
    A[4]  = (_Float16)a1.x; A[5]  = (_Float16)a1.y;
    A[6]  = (_Float16)a1.z; A[7]  = (_Float16)a1.w;
    A[8]  = (_Float16)a2.x; A[9]  = (_Float16)a2.y;
    A[10] = (_Float16)a2.z; A[11] = (_Float16)a2.w;
    A[12] = (_Float16)a3.x; A[13] = (_Float16)a3.y;
    A[14] = (_Float16)a3.z; A[15] = (_Float16)a3.w;
    return A;
}

__device__ __forceinline__ v16h load_b_frag(const _Float16* __restrict__ bp) {
    v8h b0 = *(const v8h*)(bp);
    v8h b1 = *(const v8h*)(bp + 16);
    v16h B;
#pragma unroll
    for (int e = 0; e < 8; ++e) { B[e] = b0[e]; B[8 + e] = b1[e]; }
    return B;
}

// =====================================================================
// K2: the GEMM.  proj[b][f] = sum_k x[b][k] * normals[f][k]
// One wave owns a 32(M) x 64(N) output stripe: 8 x v8f accumulators,
// 2 A fragments sharing each of 4 B fragments -> B L2 traffic halved
// vs a 16-row tile.  Register double-buffering: next K-step's A/B loads
// are issued BEFORE the current step's 8 WMMAs, so the loadcnt wait in
// front of the WMMA group covers a full iteration of slack.
// x is converted f32->f16 in-register: exactly one HBM pass (256 MB,
// ~11.5us at 23.3 TB/s — the roofline floor for this op).
// 2 waves/block, 256 blocks.
// =====================================================================
#define GEMM_WPB 2
__global__ void __launch_bounds__(32 * GEMM_WPB)
fpp_gemm(const float* __restrict__ x,
         const _Float16* __restrict__ nf16,
         float* __restrict__ proj,
         float* __restrict__ abssum) {
    __shared__ float colsum[NUM_FACETS];

    const int t    = threadIdx.x;
    const int wave = t >> 5;
    const int lane = t & 31;
    const int m    = lane & 15;
    const int half = lane >> 4;
    const int row0 = (blockIdx.x * GEMM_WPB + wave) * 32;

    if (t < NUM_FACETS) colsum[t] = 0.0f;
    __syncthreads();

    v8f acc[2][4];
#pragma unroll
    for (int mt = 0; mt < 2; ++mt)
#pragma unroll
        for (int nt = 0; nt < 4; ++nt) acc[mt][nt] = (v8f){};

    // per-lane base pointers (include the lane-half offsets)
    const float* ar0 = x + (size_t)(row0 + m) * DIM + half * 8;
    const float* ar1 = ar0 + (size_t)16 * DIM;                 // rows +16
    const _Float16* br = nf16 + (size_t)m * DIM + half * 8;    // facet m

    // ---- prologue: load K-step 0 ----
    v16h A0 = load_a_frag(ar0);
    v16h A1 = load_a_frag(ar1);
    v16h B[4];
#pragma unroll
    for (int nt = 0; nt < 4; ++nt)
        B[nt] = load_b_frag(br + (size_t)(nt * 16) * DIM);

    for (int k0 = 0; k0 < DIM; k0 += 32) {
        const int kn = (k0 + 32) & (DIM - 1);  // wraps to 0 on last iter

        // ---- issue next step's loads first (double buffer) ----
        v16h A0n = load_a_frag(ar0 + kn);
        v16h A1n = load_a_frag(ar1 + kn);
        v16h Bn[4];
#pragma unroll
        for (int nt = 0; nt < 4; ++nt)
            Bn[nt] = load_b_frag(br + (size_t)(nt * 16) * DIM + kn);
        __builtin_prefetch(ar0 + kn + 32, 0, 1);  // global_prefetch_b8

        // ---- 8 WMMAs on the current fragments ----
#pragma unroll
        for (int nt = 0; nt < 4; ++nt) {
            acc[0][nt] = __builtin_amdgcn_wmma_f32_16x16x32_f16(
                false, A0, false, B[nt], (short)0, acc[0][nt], false, false);
            acc[1][nt] = __builtin_amdgcn_wmma_f32_16x16x32_f16(
                false, A1, false, B[nt], (short)0, acc[1][nt], false, false);
        }

        A0 = A0n; A1 = A1n;
#pragma unroll
        for (int nt = 0; nt < 4; ++nt) B[nt] = Bn[nt];
    }

    // ---- store proj and fold |proj| into per-facet column sums ----
#pragma unroll
    for (int mt = 0; mt < 2; ++mt) {
#pragma unroll
        for (int nt = 0; nt < 4; ++nt) {
            float local_abs = 0.0f;
#pragma unroll
            for (int r = 0; r < 8; ++r) {
                const int row = row0 + mt * 16 + half * 8 + r;
                const int col = nt * 16 + m;
                proj[(size_t)row * NUM_FACETS + col] = acc[mt][nt][r];
                local_abs += fabsf(acc[mt][nt][r]);
            }
            atomicAdd(&colsum[nt * 16 + m], local_abs);  // ds_add_f32
        }
    }
    __syncthreads();
    if (t < NUM_FACETS) atomicAdd(&abssum[t], colsum[t]);
}

// =====================================================================
// K3: top-16 selection (matches jax.lax.top_k: descending, ties -> lower
// index first).  64 values -> trivial serial selection in one thread.
// Also writes top_idx into the output tail (int32 bits, third tuple
// member of the reference return).
// =====================================================================
__global__ void fpp_topk(const float* __restrict__ abssum,
                         const float* __restrict__ trust,
                         int* __restrict__ topidx,
                         float* __restrict__ deltas,
                         float* __restrict__ out) {
    if (threadIdx.x != 0) return;
    float mags[NUM_FACETS];
    bool  used[NUM_FACETS];
    for (int f = 0; f < NUM_FACETS; ++f) {
        mags[f] = abssum[f] * (1.0f / (float)BATCH);
        used[f] = false;
    }
    int* out_idx = (int*)(out + OUT_IDX_OFF);
    for (int j = 0; j < TOPK; ++j) {
        int   best   = -1;
        float best_v = -INFINITY;
        for (int f = 0; f < NUM_FACETS; ++f)
            if (!used[f] && mags[f] > best_v) { best_v = mags[f]; best = f; }
        used[best] = true;
        topidx[j]  = best;
        deltas[j]  = DELTA_MIN + (DELTA_MAX - DELTA_MIN) * (1.0f - trust[best]);
        out_idx[j] = best;
    }
}

// =====================================================================
// K4: nn[j] = sum_i normals[top_idx[j]][i]^2 (normals = fn / ||fn||),
// computed elementwise in f32 exactly like the reference.  16 blocks.
// =====================================================================
__global__ void fpp_nn(const float* __restrict__ fn,
                       const int* __restrict__ topidx,
                       float* __restrict__ nn) {
    __shared__ float red[256];
    const int j = blockIdx.x;
    const int t = threadIdx.x;
    const float* row = fn + (size_t)topidx[j] * DIM;

    float s = 0.0f;
    for (int i = t; i < DIM; i += 256) { float v = row[i]; s += v * v; }
    red[t] = s;
    __syncthreads();
    for (int o = 128; o > 0; o >>= 1) {
        if (t < o) red[t] += red[t + o];
        __syncthreads();
    }
    const float inv_norm = 1.0f / sqrtf(red[0]);

    s = 0.0f;
    for (int i = t; i < DIM; i += 256) {
        float v = row[i] * inv_norm;
        s += v * v;
    }
    red[t] = s;
    __syncthreads();
    for (int o = 128; o > 0; o >>= 1) {
        if (t < o) red[t] += red[t + o];
        __syncthreads();
    }
    if (t == 0) nn[j] = red[0];
}

// =====================================================================
// K5: stencil derivative + quantize + energy.
// thread id = b*16 + j.  q_all layout [order][b][j] -> coalesced writes.
// Energy reduced warp->LDS->one atomic per block.
// jnp.round == round-half-even == rintf under default RN mode.
// =====================================================================
__global__ void __launch_bounds__(256)
fpp_quantize(const float* __restrict__ proj,
             const int* __restrict__ topidx,
             const float* __restrict__ deltas,
             const float* __restrict__ nn,
             float* __restrict__ out) {
    __shared__ int   s_idx[TOPK];
    __shared__ float s_del[TOPK];
    __shared__ float s_nn[TOPK];
    __shared__ float wsum[8];

    const int t = threadIdx.x;
    if (t < TOPK) { s_idx[t] = topidx[t]; s_del[t] = deltas[t]; s_nn[t] = nn[t]; }
    __syncthreads();

    const size_t gid = (size_t)blockIdx.x * 256 + t;  // 0 .. BATCH*TOPK-1
    const int b = (int)(gid >> 4);
    const int j = (int)(gid & 15);

    const float p   = proj[(size_t)b * NUM_FACETS + s_idx[j]];
    const float nnj = s_nn[j];
    const float dj  = s_del[j];
    const float inv_dj = 1.0f / dj;

    float e = 0.0f;
    // order 1: s = {-1, 1}, offsets = {-H, 0}
    {
        float d = (p + (-H_STEP) * nnj) * (-1.0f) + (p + 0.0f * nnj) * (1.0f);
        d *= (1.0f / H_STEP);
        float q = rintf(d * inv_dj) * dj;
        out[0 * (size_t)(BATCH * TOPK) + gid] = q;
        e += fabsf(q);
    }
    // order 2: s = {1, -2, 1}, offsets = {-H, 0, H}
    {
        float d = (p + (-H_STEP) * nnj) * (1.0f)
                + (p + 0.0f * nnj)      * (-2.0f)
                + (p + H_STEP * nnj)    * (1.0f);
        d *= (1.0f / (H_STEP * H_STEP));
        float q = rintf(d * inv_dj) * dj;
        out[1 * (size_t)(BATCH * TOPK) + gid] = q;
        e += fabsf(q);
    }
    // order 3: s = {-1, 3, -3, 1}, offsets = {-2H, -H, 0, H}
    {
        float d = (p + (-2.0f * H_STEP) * nnj) * (-1.0f)
                + (p + (-H_STEP) * nnj)        * (3.0f)
                + (p + 0.0f * nnj)             * (-3.0f)
                + (p + H_STEP * nnj)           * (1.0f);
        d *= (1.0f / (H_STEP * H_STEP * H_STEP));
        float q = rintf(d * inv_dj) * dj;
        out[2 * (size_t)(BATCH * TOPK) + gid] = q;
        e += fabsf(q);
    }

    // energy reduction: warp shuffle -> LDS -> one global atomic
#pragma unroll
    for (int o = 16; o > 0; o >>= 1) e += __shfl_xor(e, o);
    if ((t & 31) == 0) wsum[t >> 5] = e;
    __syncthreads();
    if (t == 0) {
        float blk = 0.0f;
#pragma unroll
        for (int w = 0; w < 8; ++w) blk += wsum[w];
        atomicAdd(out + OUT_E_OFF, blk);
    }
}

// =====================================================================
// launch
// =====================================================================
extern "C" void kernel_launch(void* const* d_in, const int* in_sizes, int n_in,
                              void* d_out, int out_size, void* d_ws, size_t ws_size,
                              hipStream_t stream) {
    (void)in_sizes; (void)n_in; (void)out_size; (void)ws_size;

    const float* x     = (const float*)d_in[0];  // (16384, 4096)
    const float* fn    = (const float*)d_in[1];  // (64, 4096)
    const float* trust = (const float*)d_in[2];  // (64,)

    char* ws = (char*)d_ws;
    _Float16* nf16  = (_Float16*)(ws + WS_NF16);
    float*    proj  = (float*)(ws + WS_PROJ);
    float*    asum  = (float*)(ws + WS_ABSSUM);
    int*      tidx  = (int*)(ws + WS_TOPIDX);
    float*    delt  = (float*)(ws + WS_DELTAS);
    float*    nn    = (float*)(ws + WS_NN);
    float*    out   = (float*)d_out;

    fpp_zero<<<1, 128, 0, stream>>>(asum, out + OUT_E_OFF);
    fpp_normalize<<<NUM_FACETS, 256, 0, stream>>>(fn, nf16);
    fpp_gemm<<<BATCH / (32 * GEMM_WPB), 32 * GEMM_WPB, 0, stream>>>(x, nf16, proj, asum);
    fpp_topk<<<1, 32, 0, stream>>>(asum, trust, tidx, delt, out);
    fpp_nn<<<TOPK, 256, 0, stream>>>(fn, tidx, nn);
    fpp_quantize<<<(BATCH * TOPK) / 256, 256, 0, stream>>>(proj, tidx, delt, nn, out);
}
